// FusedEvoformerBlock_26620207300650
// MI455X (gfx1250) — compile-verified, matrix operand order checked by reading, not run on hardware
//
#include <hip/hip_runtime.h>

// ---------------------------------------------------------------------------
// Types for CDNA5 WMMA (wave32)
// ---------------------------------------------------------------------------
typedef __attribute__((ext_vector_type(16))) __bf16 v16bf;
typedef __attribute__((ext_vector_type(8)))  __bf16 bf16x8;
typedef __attribute__((ext_vector_type(2)))  __bf16 bf16x2;
typedef __attribute__((ext_vector_type(8)))  float  v8f;

#define EVO_S 64
#define EVO_L 256
#define EVO_D 256
#define EVO_P 128
#define EVO_HM 8
#define EVO_HP 4
#define EVO_HD 32
#define EVO_CMSA 1024
#define EVO_CPAIR 512
#define EVO_COPM 32
#define EVO_CTRI 128

__device__ __forceinline__ float sigmoidf_(float x) { return 1.0f / (1.0f + __expf(-x)); }

__device__ __forceinline__ v16bf ld_frag(const __bf16* p) {
  bf16x8 lo = *(const bf16x8*)p;
  bf16x8 hi = *(const bf16x8*)(p + 16);
  return __builtin_shufflevector(lo, hi, 0, 1, 2, 3, 4, 5, 6, 7, 8, 9, 10, 11, 12, 13, 14, 15);
}

__device__ __forceinline__ v8f wmma_bf16(v16bf a, v16bf b, v8f c) {
  return __builtin_amdgcn_wmma_f32_16x16x32_bf16(false, a, false, b, (short)0, c, false, false);
}

// ---------------------------------------------------------------------------
// Generic batched strided GEMM:  C[b] (+)= act( alpha * A[b] @ B[b] )
//   A[m][k] = A[offA + m*lda + k*inca]
//   B[k][n] = B[offB + k*ldbk + n*incbn]
//   C[m][n] = C[offC + m*ldc + n*incc]
//   batch = b1*nb2 + b2 ; off* = b1*s*1 + b2*s*2
// f32 in / f32 out, bf16 WMMA compute with f32 accumulation.
// Wave tile 32x32 (2 A-frags x 2 B-frags -> 4 accumulators): one ds_load_b128
// pair per v_wmma. KT=64 -> 8 WMMA per staging round. Uniform in-bounds test
// selects a guard-free staging path (edge tiles take the clamped path).
// ---------------------------------------------------------------------------
template <int BM, int BN, int WR, int WC>
__launch_bounds__(WR* WC * 32) __global__
void wmma_gemm_kernel(const float* __restrict__ A, const float* __restrict__ B,
                      float* __restrict__ C, int M, int N, int K,
                      int lda, int inca, int ldbk, int incbn, int ldc, int incc,
                      int tilesM, int nbatch, int nb2,
                      long long sA1, long long sA2, long long sB1, long long sB2,
                      long long sC1, long long sC2,
                      float alpha, int act, int accum) {
  constexpr int KT    = 64;
  constexpr int NT    = WR * WC * 32;
  constexpr int PPR   = KT / 2;        // bf16 pairs per staged row
  constexpr int MSTEP = NT / PPR;      // rows staged per unroll step
  static_assert(BM == WR * 32 && BN == WC * 32, "wave tile is 32x32");
  static_assert(BM % MSTEP == 0 && BN % MSTEP == 0, "tile/step mismatch");

  const int tid  = threadIdx.x;
  const int tile = blockIdx.x;
  const int tm   = tile % tilesM;
  const int tn   = tile / tilesM;
  const int batch = (int)(blockIdx.y + blockIdx.z * gridDim.y);
  if (batch >= nbatch) return;
  const int b1 = batch / nb2;
  const int b2 = batch % nb2;
  const long long offA = (long long)b1 * sA1 + (long long)b2 * sA2;
  const long long offB = (long long)b1 * sB1 + (long long)b2 * sB2;
  const long long offC = (long long)b1 * sC1 + (long long)b2 * sC2;
  const int tileM = tm * BM;
  const int tileN = tn * BN;

  // +8 pad: 144B row stride (16B aligned), staggers banks
  __shared__ __align__(16) __bf16 As[BM][KT + 8];
  __shared__ __align__(16) __bf16 Bs[BN][KT + 8];  // transposed: Bs[n][k]

  const int lane = tid & 31;
  const int wave = tid >> 5;
  const int wr   = wave % WR;
  const int wc   = wave / WR;
  const int mrow = lane & 15;
  const int half = lane >> 4;

  // staging decomposition: thread -> (row, even k)
  const int sm0 = tid / PPR;
  const int sk0 = (tid % PPR) * 2;

  // uniform fast-path test: whole macro tile and all K chunks in range
  const bool inb = (tileM + BM <= M) && (tileN + BN <= N) && (K % KT == 0);

  v8f acc00 = {}, acc01 = {}, acc10 = {}, acc11 = {};

  for (int kt = 0; kt < K; kt += KT) {
    if (inb) {
      // ---- guard-free staging ----
#pragma unroll
      for (int it = 0; it < BM / MSTEP; ++it) {
        const int m = sm0 + it * MSTEP;
        const long long rowo =
            offA + (long long)(tileM + m) * lda + (long long)(kt + sk0) * inca;
        float v0 = A[rowo];
        float v1 = A[rowo + inca];
        bf16x2 pk;
        pk.x = (__bf16)v0;
        pk.y = (__bf16)v1;
        *(bf16x2*)&As[m][sk0] = pk;
      }
#pragma unroll
      for (int it = 0; it < BN / MSTEP; ++it) {
        const int n = sm0 + it * MSTEP;
        const long long colo =
            offB + (long long)(tileN + n) * incbn + (long long)(kt + sk0) * ldbk;
        float v0 = B[colo];
        float v1 = B[colo + ldbk];
        bf16x2 pk;
        pk.x = (__bf16)v0;
        pk.y = (__bf16)v1;
        *(bf16x2*)&Bs[n][sk0] = pk;
      }
    } else {
      // ---- clamped staging for edge tiles ----
#pragma unroll
      for (int it = 0; it < BM / MSTEP; ++it) {
        const int m  = sm0 + it * MSTEP;
        const int gm = tileM + m;
        const int gk0 = kt + sk0;
        const int gk1 = gk0 + 1;
        const int cm  = gm  < M ? gm  : M - 1;
        const int ck0 = gk0 < K ? gk0 : K - 1;
        const int ck1 = gk1 < K ? gk1 : K - 1;
        const long long rowo = offA + (long long)cm * lda;
        float v0 = A[rowo + (long long)ck0 * inca];
        float v1 = A[rowo + (long long)ck1 * inca];
        if (!(gm < M && gk0 < K)) v0 = 0.0f;
        if (!(gm < M && gk1 < K)) v1 = 0.0f;
        bf16x2 pk;
        pk.x = (__bf16)v0;
        pk.y = (__bf16)v1;
        *(bf16x2*)&As[m][sk0] = pk;
      }
#pragma unroll
      for (int it = 0; it < BN / MSTEP; ++it) {
        const int n  = sm0 + it * MSTEP;
        const int gn = tileN + n;
        const int gk0 = kt + sk0;
        const int gk1 = gk0 + 1;
        const int cn  = gn  < N ? gn  : N - 1;
        const int ck0 = gk0 < K ? gk0 : K - 1;
        const int ck1 = gk1 < K ? gk1 : K - 1;
        const long long colo = offB + (long long)cn * incbn;
        float v0 = B[colo + (long long)ck0 * ldbk];
        float v1 = B[colo + (long long)ck1 * ldbk];
        if (!(gn < N && gk0 < K)) v0 = 0.0f;
        if (!(gn < N && gk1 < K)) v1 = 0.0f;
        bf16x2 pk;
        pk.x = (__bf16)v0;
        pk.y = (__bf16)v1;
        *(bf16x2*)&Bs[n][sk0] = pk;
      }
    }
    // prefetch next K chunk (global_prefetch_b8)
    if (kt + KT < K) {
      __builtin_prefetch((const void*)(A + offA + (long long)(tileM + sm0) * lda +
                                       (long long)(kt + KT + sk0) * inca), 0, 3);
      __builtin_prefetch((const void*)(B + offB + (long long)(tileN + (sm0 & (BN - 1))) * incbn +
                                       (long long)(kt + KT + sk0) * ldbk), 0, 3);
    }
    __syncthreads();

    // ---- 2 K-chunks x (2x2) wave tile = 8 WMMA per round ----
#pragma unroll
    for (int kk = 0; kk < KT; kk += 32) {
      v16bf a0 = ld_frag(&As[wr * 32 + mrow][kk + half * 8]);
      v16bf a1 = ld_frag(&As[wr * 32 + 16 + mrow][kk + half * 8]);
      v16bf b0 = ld_frag(&Bs[wc * 32 + mrow][kk + half * 8]);
      v16bf b1 = ld_frag(&Bs[wc * 32 + 16 + mrow][kk + half * 8]);
      acc00 = wmma_bf16(a0, b0, acc00);
      acc01 = wmma_bf16(a0, b1, acc01);
      acc10 = wmma_bf16(a1, b0, acc10);
      acc11 = wmma_bf16(a1, b1, acc11);
    }
    __syncthreads();
  }

  // ---- epilogue: D layout M = (lane>>4)*8 + r, N = lane&15 ----
  const int n0 = tileN + wc * 32 + mrow;
  const int n1 = n0 + 16;
#pragma unroll
  for (int half_m = 0; half_m < 2; ++half_m) {
    const v8f& ac0 = half_m ? acc10 : acc00;
    const v8f& ac1 = half_m ? acc11 : acc01;
#pragma unroll
    for (int r = 0; r < 8; ++r) {
      const int m = tileM + wr * 32 + half_m * 16 + half * 8 + r;
      if (m < M) {
        const long long rowc = offC + (long long)m * ldc;
        if (n0 < N) {
          float v = alpha * ac0[r];
          if (act == 1) v = v > 0.0f ? v : 0.0f;
          else if (act == 2) v = sigmoidf_(v);
          long long off = rowc + (long long)n0 * incc;
          if (accum) C[off] += v; else C[off] = v;
        }
        if (n1 < N) {
          float v = alpha * ac1[r];
          if (act == 1) v = v > 0.0f ? v : 0.0f;
          else if (act == 2) v = sigmoidf_(v);
          long long off = rowc + (long long)n1 * incc;
          if (accum) C[off] += v; else C[off] = v;
        }
      }
    }
  }
}

// ---------------------------------------------------------------------------
// LayerNorm over last dim (one block per row)
// ---------------------------------------------------------------------------
__global__ void ln_kernel(const float* __restrict__ x, const float* __restrict__ g,
                          const float* __restrict__ b, float* __restrict__ y,
                          int rows, int dim) {
  int row = blockIdx.x;
  if (row >= rows) return;
  const float* xr = x + (long long)row * dim;
  float* yr = y + (long long)row * dim;
  __shared__ float r1[128], r2[128];
  int tid = threadIdx.x;
  float s = 0.0f, s2 = 0.0f;
  for (int k = tid; k < dim; k += 128) { float v = xr[k]; s += v; s2 += v * v; }
  r1[tid] = s; r2[tid] = s2; __syncthreads();
  for (int st = 64; st > 0; st >>= 1) {
    if (tid < st) { r1[tid] += r1[tid + st]; r2[tid] += r2[tid + st]; }
    __syncthreads();
  }
  float mu  = r1[0] / dim;
  float var = r2[0] / dim - mu * mu;
  float inv = rsqrtf(var + 1e-5f);
  for (int k = tid; k < dim; k += 128) yr[k] = (xr[k] - mu) * inv * g[k] + b[k];
}

// ---------------------------------------------------------------------------
// Softmax over last dim, in place, optional bias with generic indexing:
//   bias_idx(k) = ((row/divA)%modA)*strA + ((row/divB)%modB)*strB + k*bk
// ---------------------------------------------------------------------------
__global__ void softmax_kernel(float* __restrict__ x, int rows, int len,
                               const float* __restrict__ bias, int bk,
                               int divA, int modA, long long strA,
                               int divB, int modB, long long strB) {
  int row = blockIdx.x;
  if (row >= rows) return;
  float* xr = x + (long long)row * len;
  const bool hasb = bias != nullptr;
  long long bbase = 0;
  if (hasb)
    bbase = (long long)((row / divA) % modA) * strA + (long long)((row / divB) % modB) * strB;
  __shared__ float red[128];
  int tid = threadIdx.x;
  float mx = -3.4e38f;
  for (int k = tid; k < len; k += 128) {
    float v = xr[k];
    if (hasb) v += bias[bbase + (long long)k * bk];
    xr[k] = v;
    mx = fmaxf(mx, v);
  }
  red[tid] = mx; __syncthreads();
  for (int st = 64; st > 0; st >>= 1) {
    if (tid < st) red[tid] = fmaxf(red[tid], red[tid + st]);
    __syncthreads();
  }
  mx = red[0]; __syncthreads();
  float sum = 0.0f;
  for (int k = tid; k < len; k += 128) { float e = __expf(xr[k] - mx); xr[k] = e; sum += e; }
  red[tid] = sum; __syncthreads();
  for (int st = 64; st > 0; st >>= 1) {
    if (tid < st) red[tid] += red[tid + st];
    __syncthreads();
  }
  float inv = 1.0f / red[0];
  for (int k = tid; k < len; k += 128) xr[k] *= inv;
}

// ---------------------------------------------------------------------------
// Elementwise helpers
// ---------------------------------------------------------------------------
__global__ void gate_mul_kernel(float* __restrict__ x, const float* __restrict__ g, long long n) {
  long long i  = (long long)blockIdx.x * blockDim.x + threadIdx.x;
  long long st = (long long)gridDim.x * blockDim.x;
  for (; i < n; i += st) x[i] *= sigmoidf_(g[i]);
}

__global__ void gated_residual_kernel(float* __restrict__ out, const float* __restrict__ g,
                                      const float* __restrict__ o, long long n) {
  long long i  = (long long)blockIdx.x * blockDim.x + threadIdx.x;
  long long st = (long long)gridDim.x * blockDim.x;
  for (; i < n; i += st) out[i] += sigmoidf_(g[i]) * o[i];
}

__global__ void transpose_add_kernel(float* __restrict__ out, const float* __restrict__ t,
                                     int Ldim, int Pdim, long long n) {
  long long i  = (long long)blockIdx.x * blockDim.x + threadIdx.x;
  long long st = (long long)gridDim.x * blockDim.x;
  for (; i < n; i += st) {
    int c = (int)(i % Pdim);
    long long r = i / Pdim;
    int jj = (int)(r % Ldim);
    int ii = (int)(r / Ldim);
    out[i] += t[((long long)jj * Ldim + ii) * Pdim + c];
  }
}

// ---------------------------------------------------------------------------
// Host-side launch helpers
// ---------------------------------------------------------------------------
static void gemm(hipStream_t st, bool small, const float* A, const float* B, float* C,
                 int M, int N, int K,
                 int lda, int inca, int ldbk, int incbn, int ldc, int incc,
                 int nbatch, int nb2,
                 long long sA1, long long sA2, long long sB1, long long sB2,
                 long long sC1, long long sC2,
                 float alpha, int act, int accum) {
  int BM = small ? 32 : 64;
  int BN = small ? 32 : 64;
  int tilesM = (M + BM - 1) / BM;
  int tilesN = (N + BN - 1) / BN;
  int gy = nbatch, gz = 1;
  if (gy > 32768) { gy = 32768; gz = (nbatch + 32767) / 32768; }
  dim3 grid((unsigned)(tilesM * tilesN), (unsigned)gy, (unsigned)gz);
  if (small)
    wmma_gemm_kernel<32, 32, 1, 1><<<grid, 32, 0, st>>>(
        A, B, C, M, N, K, lda, inca, ldbk, incbn, ldc, incc, tilesM, nbatch, nb2,
        sA1, sA2, sB1, sB2, sC1, sC2, alpha, act, accum);
  else
    wmma_gemm_kernel<64, 64, 2, 2><<<grid, 128, 0, st>>>(
        A, B, C, M, N, K, lda, inca, ldbk, incbn, ldc, incc, tilesM, nbatch, nb2,
        sA1, sA2, sB1, sB2, sC1, sC2, alpha, act, accum);
}

static void run_ln(hipStream_t st, const float* x, const float* g, const float* b,
                   float* y, int rows, int dim) {
  ln_kernel<<<rows, 128, 0, st>>>(x, g, b, y, rows, dim);
}

static void run_elem(hipStream_t st, int which, float* a, const float* b, const float* c,
                     long long n) {
  int blocks = (int)((n + 255) / 256);
  if (blocks > 32768) blocks = 32768;
  if (which == 0)      gate_mul_kernel<<<blocks, 256, 0, st>>>(a, b, n);
  else if (which == 1) gated_residual_kernel<<<blocks, 256, 0, st>>>(a, b, c, n);
}

// scores/softmax/AV shared by triangle attention (tas / tae) -- qkv already in (i,j) frame
static void tri_attn_core(hipStream_t st, float* qkvb, float* biasb, float* sb, float* ob) {
  const float sc = 0.17677669529663687f;  // 1/sqrt(32)
  // scores[h,i,j,k] = sum_d q[i,j,h,d] * k[i,k,h,d]  (batch = h*L + i)
  gemm(st, false, qkvb, qkvb + EVO_P, sb, EVO_L, EVO_L, EVO_HD,
       3 * EVO_P, 1, 1, 3 * EVO_P, EVO_L, 1,
       EVO_HP * EVO_L, EVO_L,
       32, (long long)EVO_L * 3 * EVO_P, 32, (long long)EVO_L * 3 * EVO_P,
       (long long)EVO_L * EVO_L * EVO_L, (long long)EVO_L * EVO_L,
       sc, 0, 0);
  // softmax over k with bias[h,j,k] stored as (j,k,h)
  softmax_kernel<<<EVO_HP * EVO_L * EVO_L, 128, 0, st>>>(
      sb, EVO_HP * EVO_L * EVO_L, EVO_L, biasb, EVO_HP,
      1, EVO_L, (long long)EVO_L * EVO_HP, EVO_L * EVO_L, EVO_HP, 1LL);
  // o[i,j,h,d] = sum_k a[h,i,j,k] * v[i,k,h,d]
  gemm(st, false, sb, qkvb + 2 * EVO_P, ob, EVO_L, EVO_HD, EVO_L,
       EVO_L, 1, 3 * EVO_P, 1, EVO_P, 1,
       EVO_HP * EVO_L, EVO_L,
       (long long)EVO_L * EVO_L * EVO_L, (long long)EVO_L * EVO_L,
       32, (long long)EVO_L * 3 * EVO_P,
       32, (long long)EVO_L * EVO_P,
       1.0f, 0, 0);
}

static void tri_mult(hipStream_t st, float* pairO, float* lnb, float* ab, float* bb,
                     float* gb, float* sb,
                     const float* g_ln, const float* b_ln,
                     const float* g_ln2, const float* b_ln2,
                     const float* wa, const float* wga, const float* wb, const float* wgb,
                     const float* wo, const float* wg, bool outgoing) {
  const int LL = EVO_L * EVO_L;          // 65536
  const long long NP = (long long)LL * EVO_P;
  run_ln(st, pairO, g_ln, b_ln, lnb, LL, EVO_P);
  gemm(st, false, lnb, wa, ab, LL, EVO_CTRI, EVO_P, EVO_P, 1, EVO_CTRI, 1, EVO_CTRI, 1,
       1, 1, 0, 0, 0, 0, 0, 0, 1.0f, 0, 0);
  gemm(st, false, lnb, wga, gb, LL, EVO_CTRI, EVO_P, EVO_P, 1, EVO_CTRI, 1, EVO_CTRI, 1,
       1, 1, 0, 0, 0, 0, 0, 0, 1.0f, 0, 0);
  run_elem(st, 0, ab, gb, nullptr, NP);
  gemm(st, false, lnb, wb, bb, LL, EVO_CTRI, EVO_P, EVO_P, 1, EVO_CTRI, 1, EVO_CTRI, 1,
       1, 1, 0, 0, 0, 0, 0, 0, 1.0f, 0, 0);
  gemm(st, false, lnb, wgb, gb, LL, EVO_CTRI, EVO_P, EVO_P, 1, EVO_CTRI, 1, EVO_CTRI, 1,
       1, 1, 0, 0, 0, 0, 0, 0, 1.0f, 0, 0);
  run_elem(st, 0, bb, gb, nullptr, NP);
  // per-channel triangle einsum (batch c over C_TRI)
  const int LC = EVO_L * EVO_CTRI;       // 32768
  if (outgoing)  // O[i,j,c] = sum_k a[i,k,c] b[j,k,c]
    gemm(st, false, ab, bb, sb, EVO_L, EVO_L, EVO_L,
         LC, EVO_CTRI, EVO_CTRI, LC, LC, EVO_CTRI,
         EVO_CTRI, EVO_CTRI, 0, 1, 0, 1, 0, 1, 1.0f, 0, 0);
  else           // O[i,j,c] = sum_k a[k,i,c] b[k,j,c]
    gemm(st, false, ab, bb, sb, EVO_L, EVO_L, EVO_L,
         EVO_CTRI, LC, LC, EVO_CTRI, LC, EVO_CTRI,
         EVO_CTRI, EVO_CTRI, 0, 1, 0, 1, 0, 1, 1.0f, 0, 0);
  run_ln(st, sb, g_ln2, b_ln2, gb, LL, EVO_CTRI);
  gemm(st, false, gb, wo, ab, LL, EVO_P, EVO_CTRI, EVO_CTRI, 1, EVO_P, 1, EVO_P, 1,
       1, 1, 0, 0, 0, 0, 0, 0, 1.0f, 0, 0);
  gemm(st, false, lnb, wg, bb, LL, EVO_P, EVO_P, EVO_P, 1, EVO_P, 1, EVO_P, 1,
       1, 1, 0, 0, 0, 0, 0, 0, 1.0f, 0, 0);
  run_elem(st, 1, pairO, bb, ab, NP);    // pairO += sigmoid(gate) * proj
}

// ---------------------------------------------------------------------------
// kernel_launch: full Evoformer block
// ---------------------------------------------------------------------------
extern "C" void kernel_launch(void* const* d_in, const int* in_sizes, int n_in,
                              void* d_out, int out_size, void* d_ws, size_t ws_size,
                              hipStream_t stream) {
  (void)in_sizes; (void)n_in; (void)out_size; (void)ws_size;
  auto IN = [&](int i) { return (const float*)d_in[i]; };

  const int SL  = EVO_S * EVO_L;         // 16384
  const int LL  = EVO_L * EVO_L;         // 65536
  const long long msaN  = (long long)SL * EVO_D;   // 4,194,304
  const long long pairN = (long long)LL * EVO_P;   // 8,388,608

  float* msaO  = (float*)d_out;
  float* pairO = msaO + msaN;

  // workspace arena
  float* p = (float*)d_ws;
  float* lnbuf  = p; p += 8388608;                 // layernorm outputs
  float* qkvbuf = p; p += 25165824;                // up to (65536,384)
  float* sbuf   = p; p += 67108864;                // scores / OPM O / transition hidden
  float* obuf   = p; p += 8388608;                 // attention outputs
  float* abuf   = p; p += 8388608;
  float* bbuf2  = p; p += 8388608;
  float* gbuf   = p; p += 8388608;
  float* tbuf   = p; p += 8388608;
  float* biasb  = p; p += 524288;                  // attention bias matrices

  const float sc32 = 0.17677669529663687f;         // 1/sqrt(32)

  // initialize residual streams
  hipMemcpyAsync(msaO,  d_in[0], msaN  * sizeof(float), hipMemcpyDeviceToDevice, stream);
  hipMemcpyAsync(pairO, d_in[1], pairN * sizeof(float), hipMemcpyDeviceToDevice, stream);

  // ========== 1) MSA row attention with pair bias ==========
  run_ln(stream, msaO, IN(2), IN(3), lnbuf, SL, EVO_D);
  gemm(stream, false, lnbuf, IN(6), qkvbuf, SL, 3 * EVO_D, EVO_D,
       EVO_D, 1, 3 * EVO_D, 1, 3 * EVO_D, 1, 1, 1, 0, 0, 0, 0, 0, 0, 1.0f, 0, 0);
  run_ln(stream, pairO, IN(4), IN(5), lnbuf, LL, EVO_P);
  gemm(stream, false, lnbuf, IN(7), biasb, LL, EVO_HM, EVO_P,
       EVO_P, 1, EVO_HM, 1, EVO_HM, 1, 1, 1, 0, 0, 0, 0, 0, 0, 1.0f, 0, 0);
  // scores[s,h,i,j] = q . k  (batch = s*H + h)
  gemm(stream, false, qkvbuf, qkvbuf + EVO_D, sbuf, EVO_L, EVO_L, EVO_HD,
       3 * EVO_D, 1, 1, 3 * EVO_D, EVO_L, 1,
       EVO_S * EVO_HM, EVO_HM,
       (long long)EVO_L * 3 * EVO_D, 32, (long long)EVO_L * 3 * EVO_D, 32,
       (long long)EVO_HM * LL, (long long)LL, sc32, 0, 0);
  softmax_kernel<<<EVO_S * EVO_HM * EVO_L, 128, 0, stream>>>(
      sbuf, EVO_S * EVO_HM * EVO_L, EVO_L, biasb, EVO_HM,
      1, EVO_L, (long long)EVO_L * EVO_HM, EVO_L, EVO_HM, 1LL);
  // o[s,i,h,d] = sum_j a[s,h,i,j] v[s,j,h,d]
  gemm(stream, false, sbuf, qkvbuf + 2 * EVO_D, obuf, EVO_L, EVO_HD, EVO_L,
       EVO_L, 1, 3 * EVO_D, 1, EVO_D, 1,
       EVO_S * EVO_HM, EVO_HM,
       (long long)EVO_HM * LL, (long long)LL,
       (long long)EVO_L * 3 * EVO_D, 32,
       (long long)EVO_L * EVO_D, 32, 1.0f, 0, 0);
  gemm(stream, false, obuf, IN(8), msaO, SL, EVO_D, EVO_D,
       EVO_D, 1, EVO_D, 1, EVO_D, 1, 1, 1, 0, 0, 0, 0, 0, 0, 1.0f, 0, 1);

  // ========== 2) MSA column attention ==========
  run_ln(stream, msaO, IN(9), IN(10), lnbuf, SL, EVO_D);
  gemm(stream, false, lnbuf, IN(11), qkvbuf, SL, 3 * EVO_D, EVO_D,
       EVO_D, 1, 3 * EVO_D, 1, 3 * EVO_D, 1, 1, 1, 0, 0, 0, 0, 0, 0, 1.0f, 0, 0);
  // scores[i,h,s,t]  (batch = i*H + h)
  gemm(stream, false, qkvbuf, qkvbuf + EVO_D, sbuf, EVO_S, EVO_S, EVO_HD,
       (long long)EVO_L * 3 * EVO_D, 1, 1, EVO_L * 3 * EVO_D, EVO_S, 1,
       EVO_L * EVO_HM, EVO_HM,
       3 * EVO_D, 32, 3 * EVO_D, 32,
       (long long)EVO_HM * EVO_S * EVO_S, (long long)EVO_S * EVO_S, sc32, 0, 0);
  softmax_kernel<<<EVO_L * EVO_HM * EVO_S, 128, 0, stream>>>(
      sbuf, EVO_L * EVO_HM * EVO_S, EVO_S, nullptr, 1, 1, 1, 0LL, 1, 1, 0LL);
  // o[s,i,h,d] = sum_t a[i,h,s,t] v[t,i,h,d]
  gemm(stream, false, sbuf, qkvbuf + 2 * EVO_D, obuf, EVO_S, EVO_HD, EVO_S,
       EVO_S, 1, EVO_L * 3 * EVO_D, 1, EVO_L * EVO_D, 1,
       EVO_L * EVO_HM, EVO_HM,
       (long long)EVO_HM * EVO_S * EVO_S, (long long)EVO_S * EVO_S,
       3 * EVO_D, 32, EVO_D, 32, 1.0f, 0, 0);
  gemm(stream, false, obuf, IN(12), msaO, SL, EVO_D, EVO_D,
       EVO_D, 1, EVO_D, 1, EVO_D, 1, 1, 1, 0, 0, 0, 0, 0, 0, 1.0f, 0, 1);

  // ========== 3) MSA transition ==========
  run_ln(stream, msaO, IN(13), IN(14), lnbuf, SL, EVO_D);
  gemm(stream, false, lnbuf, IN(15), sbuf, SL, EVO_CMSA, EVO_D,
       EVO_D, 1, EVO_CMSA, 1, EVO_CMSA, 1, 1, 1, 0, 0, 0, 0, 0, 0, 1.0f, 1, 0);
  gemm(stream, false, sbuf, IN(16), msaO, SL, EVO_D, EVO_CMSA,
       EVO_CMSA, 1, EVO_D, 1, EVO_D, 1, 1, 1, 0, 0, 0, 0, 0, 0, 1.0f, 0, 1);

  // ========== 4) Outer product mean ==========
  run_ln(stream, msaO, IN(17), IN(18), lnbuf, SL, EVO_D);
  gemm(stream, false, lnbuf, IN(19), abuf, SL, EVO_COPM, EVO_D,
       EVO_D, 1, EVO_COPM, 1, EVO_COPM, 1, 1, 1, 0, 0, 0, 0, 0, 0, 1.0f, 0, 0);
  gemm(stream, false, lnbuf, IN(20), bbuf2, SL, EVO_COPM, EVO_D,
       EVO_D, 1, EVO_COPM, 1, EVO_COPM, 1, 1, 1, 0, 0, 0, 0, 0, 0, 1.0f, 0, 0);
  // O[i,j,c,d] = (1/S) sum_s a[s,i,c] b[s,j,d]  (batch = i*L + j, 32x32x64)
  gemm(stream, true, abuf, bbuf2, sbuf, EVO_COPM, EVO_COPM, EVO_S,
       1, EVO_L * EVO_COPM, EVO_L * EVO_COPM, 1, EVO_COPM, 1,
       LL, EVO_L,
       EVO_COPM, 0, 0, EVO_COPM,
       (long long)EVO_L * EVO_COPM * EVO_COPM, (long long)EVO_COPM * EVO_COPM,
       1.0f / EVO_S, 0, 0);
  gemm(stream, false, sbuf, IN(21), pairO, LL, EVO_P, EVO_COPM * EVO_COPM,
       EVO_COPM * EVO_COPM, 1, EVO_P, 1, EVO_P, 1, 1, 1, 0, 0, 0, 0, 0, 0, 1.0f, 0, 1);

  // ========== 5/6) Triangle multiplicative updates ==========
  tri_mult(stream, pairO, lnbuf, abuf, bbuf2, gbuf, sbuf,
           IN(22), IN(23), IN(24), IN(25), IN(26), IN(27), IN(28), IN(29), IN(30), IN(31),
           true);   // outgoing
  tri_mult(stream, pairO, lnbuf, abuf, bbuf2, gbuf, sbuf,
           IN(32), IN(33), IN(34), IN(35), IN(36), IN(37), IN(38), IN(39), IN(40), IN(41),
           false);  // incoming

  // ========== 7) Triangle attention (starting node) ==========
  run_ln(stream, pairO, IN(42), IN(43), lnbuf, LL, EVO_P);
  gemm(stream, false, lnbuf, IN(44), qkvbuf, LL, 3 * EVO_P, EVO_P,
       EVO_P, 1, 3 * EVO_P, 1, 3 * EVO_P, 1, 1, 1, 0, 0, 0, 0, 0, 0, 1.0f, 0, 0);
  gemm(stream, false, lnbuf, IN(45), biasb, LL, EVO_HP, EVO_P,
       EVO_P, 1, EVO_HP, 1, EVO_HP, 1, 1, 1, 0, 0, 0, 0, 0, 0, 1.0f, 0, 0);
  tri_attn_core(stream, qkvbuf, biasb, sbuf, obuf);
  gemm(stream, false, obuf, IN(46), pairO, LL, EVO_P, EVO_P,
       EVO_P, 1, EVO_P, 1, EVO_P, 1, 1, 1, 0, 0, 0, 0, 0, 0, 1.0f, 0, 1);

  // ========== 8) Triangle attention (ending node) via implicit transpose ==========
  run_ln(stream, pairO, IN(47), IN(48), lnbuf, LL, EVO_P);
  // qkvT[(i,j)] = lnbuf[(j,i)] @ tae_qkv   (batch over i, rows = j)
  gemm(stream, false, lnbuf, IN(49), qkvbuf, EVO_L, 3 * EVO_P, EVO_P,
       EVO_L * EVO_P, 1, 3 * EVO_P, 1, 3 * EVO_P, 1,
       EVO_L, EVO_L, 0, EVO_P, 0, 0, 0, (long long)EVO_L * 3 * EVO_P, 1.0f, 0, 0);
  gemm(stream, false, lnbuf, IN(50), biasb, EVO_L, EVO_HP, EVO_P,
       EVO_L * EVO_P, 1, EVO_HP, 1, EVO_HP, 1,
       EVO_L, EVO_L, 0, EVO_P, 0, 0, 0, (long long)EVO_L * EVO_HP, 1.0f, 0, 0);
  tri_attn_core(stream, qkvbuf, biasb, sbuf, obuf);
  gemm(stream, false, obuf, IN(51), tbuf, LL, EVO_P, EVO_P,
       EVO_P, 1, EVO_P, 1, EVO_P, 1, 1, 1, 0, 0, 0, 0, 0, 0, 1.0f, 0, 0);
  {
    int blocks = (int)((pairN + 255) / 256);
    if (blocks > 32768) blocks = 32768;
    transpose_add_kernel<<<blocks, 256, 0, stream>>>(pairO, tbuf, EVO_L, EVO_P, pairN);
  }

  // ========== 9) Pair transition ==========
  run_ln(stream, pairO, IN(52), IN(53), lnbuf, LL, EVO_P);
  gemm(stream, false, lnbuf, IN(54), sbuf, LL, EVO_CPAIR, EVO_P,
       EVO_P, 1, EVO_CPAIR, 1, EVO_CPAIR, 1, 1, 1, 0, 0, 0, 0, 0, 0, 1.0f, 1, 0);
  gemm(stream, false, sbuf, IN(55), pairO, LL, EVO_P, EVO_CPAIR,
       EVO_CPAIR, 1, EVO_P, 1, EVO_P, 1, 1, 1, 0, 0, 0, 0, 0, 0, 1.0f, 0, 1);
}
